// LAMMA3_84808424227286
// MI455X (gfx1250) — matrix operand order repaired and checked by reading, not compiled, not run on hardware
//
#include <hip/hip_runtime.h>
#include <math.h>

// ---------------- types / helpers ----------------
typedef __attribute__((ext_vector_type(16))) __bf16 v16bf;
typedef __attribute__((ext_vector_type(8)))  __bf16 v8bf;
typedef __attribute__((ext_vector_type(8)))  float  v8f;

#define LM_EPS 1e-5f

static __device__ __forceinline__ unsigned short f2bf(float f) {
    unsigned u = __builtin_bit_cast(unsigned, f);
    u = u + 0x7fffu + ((u >> 16) & 1u);          // round-to-nearest-even
    return (unsigned short)(u >> 16);
}

static __device__ __forceinline__ v8f wmma_bf(v16bf a, v16bf b, v8f c) {
    // v_wmma_f32_16x16x32_bf16: (neg_a, A, neg_b, B, c_mod, C, reuse_a, reuse_b)
    return __builtin_amdgcn_wmma_f32_16x16x32_bf16(false, a, false, b, (short)0, c, false, false);
}

// A fragment: 16x32 bf16 from row-major A[m][k] (lda elems). Two 16B vector loads.
// lane L: m = L&15 ; lanes0-15 K {k0..k0+15}, lanes16-31 K {k0+16..k0+31}
static __device__ __forceinline__ v16bf load_afrag(const unsigned short* A, int lda,
                                                   int m0, int k0, int lane) {
    const unsigned short* p = A + (size_t)(m0 + (lane & 15)) * lda + k0 + ((lane >> 4) << 4);
    v8bf lo = *(const v8bf*)p;
    v8bf hi = *(const v8bf*)(p + 8);
    return __builtin_shufflevector(lo, hi, 0, 1, 2, 3, 4, 5, 6, 7, 8, 9, 10, 11, 12, 13, 14, 15);
}

// swizzled-B tile index: tile = 32 lanes x 16 elems (1 KB), lane = hi16(k)*16 + (n&15), e = k&15
static __device__ __forceinline__ size_t bsw_idx(int k, int n, int N) {
    return ((size_t)((k >> 5) * (N >> 4) + (n >> 4)) * 32 + (((k >> 4) & 1) << 4) + (n & 15)) * 16
           + (k & 15);
}

// ---------------- generic GEMM: C(MxN,f32) = A(MxK,bf16 row-major) @ Bsw(KxN, swizzled) ---
// block: 256 thr = 8 waves, computes 128x64; B panels double-buffered via async-to-LDS DMA
// (ASYNCcnt completes in order -> s_wait_asynccnt 1 with two panels in flight).
__global__ __launch_bounds__(256) void lm3_gemm(const unsigned short* __restrict__ A,
                                                const unsigned short* __restrict__ Bsw,
                                                float* __restrict__ C, int M, int N, int K) {
    __shared__ __align__(32) unsigned short Bsh[4096];   // 2 x (32x64 bf16 panel)
    const int tid = threadIdx.x;
    const int wave = tid >> 5, lane = tid & 31;
    const int mr = lane & 15, hi = lane >> 4;
    const int m0 = blockIdx.x * 128 + wave * 16;
    const int n0 = blockIdx.y * 64;
    const int nt16 = N >> 4;
    const int nk = K >> 5;
    v8f acc[4];
#pragma unroll
    for (int t = 0; t < 4; ++t)
#pragma unroll
        for (int r = 0; r < 8; ++r) acc[t][r] = 0.f;
    const unsigned ldsA = (unsigned)(size_t)(&Bsh[tid * 8]);     // LDS offset = low 32 bits
    const unsigned short* gbase = Bsw + ((size_t)(blockIdx.y << 2)) * 512 + tid * 8;
    const size_t kstride = (size_t)nt16 * 512;                   // elems per k-step of B

#define LM_ISSUE(ktv)                                                                   \
    do {                                                                                \
        unsigned long long ga_ = (unsigned long long)(size_t)(gbase + (size_t)(ktv) * kstride); \
        unsigned lds_ = ldsA + (((ktv) & 1) << 12);                                     \
        asm volatile("global_load_async_to_lds_b128 %0, %1, off"                        \
                     :: "v"(lds_), "v"(ga_) : "memory");                                \
    } while (0)

    __builtin_prefetch(A + (size_t)m0 * K, 0, 1);
    LM_ISSUE(0);
    if (nk > 1) LM_ISSUE(1);
    for (int kt = 0; kt < nk; ++kt) {
        if (kt + 1 < nk) asm volatile("s_wait_asynccnt 0x1" ::: "memory");
        else             asm volatile("s_wait_asynccnt 0x0" ::: "memory");
        __syncthreads();
        v16bf a = load_afrag(A, K, m0, kt * 32, lane);
        const unsigned short* pb = &Bsh[((kt & 1) << 11) + lane * 16];
#pragma unroll
        for (int t = 0; t < 4; ++t) {
            v16bf b = *(const v16bf*)(pb + t * 512);
            acc[t] = wmma_bf(a, b, acc[t]);
        }
        __syncthreads();                    // readers done before this buffer is re-DMA'd
        if (kt + 2 < nk) LM_ISSUE(kt + 2);
    }
#undef LM_ISSUE
#pragma unroll
    for (int t = 0; t < 4; ++t) {
        int n = n0 + t * 16 + mr;
        int mb = m0 + (hi << 3);
#pragma unroll
        for (int r = 0; r < 8; ++r) C[(size_t)(mb + r) * N + n] = acc[t][r];
    }
}

// ---------------- flash attention: one wave = 16 queries of one (bn, head) --------------
// Q: bf16 row-major [bn][2048][512] (pre-scaled by DH^-0.5), head base h*128.
// Ksw: [bn*4+h][kk(4)][keytile(64)][lane][16]  (B frags of K^T)
// Vsw: [bn*4+h][kt(32)][nt(8)][lane][16]       (B frags of V)
__global__ __launch_bounds__(32) void lm3_attn(const unsigned short* __restrict__ Qb,
                                               const unsigned short* __restrict__ Ksw,
                                               const unsigned short* __restrict__ Vsw,
                                               unsigned short* __restrict__ Ob) {
    __shared__ __align__(32) unsigned short Pl[512];     // 16x32 bf16, A-frag swizzled
    const int lane = threadIdx.x;
    const int qt = blockIdx.x, h = blockIdx.y, bn = blockIdx.z;
    const int mr = lane & 15, hi = lane >> 4;
    const unsigned short* Q = Qb + (size_t)bn * 2048 * 512 + (size_t)h * 128;
    const size_t bh = (size_t)(bn * 4 + h) * 131072;
    const unsigned short* Kp = Ksw + bh;
    const unsigned short* Vp = Vsw + bh;

    v16bf qf[4];
#pragma unroll
    for (int i = 0; i < 4; ++i) qf[i] = load_afrag(Q, 512, qt * 16, i * 32, lane);

    v8f o[8];
#pragma unroll
    for (int nt = 0; nt < 8; ++nt)
#pragma unroll
        for (int r = 0; r < 8; ++r) o[nt][r] = 0.f;
    float mrun[8], lrun[8];
#pragma unroll
    for (int r = 0; r < 8; ++r) { mrun[r] = -1e30f; lrun[r] = 0.f; }

    for (int kt = 0; kt < 32; ++kt) {
        v8f s0, s1;
#pragma unroll
        for (int r = 0; r < 8; ++r) { s0[r] = 0.f; s1[r] = 0.f; }
#pragma unroll
        for (int kk = 0; kk < 4; ++kk) {
            v16bf b0 = *(const v16bf*)(Kp + (size_t)kk * 32768 + (size_t)(2 * kt) * 512 + lane * 16);
            v16bf b1 = *(const v16bf*)(Kp + (size_t)kk * 32768 + (size_t)(2 * kt + 1) * 512 + lane * 16);
            s0 = wmma_bf(qf[kk], b0, s0);
            s1 = wmma_bf(qf[kk], b1, s1);
        }
        // online softmax; row = r + 8*hi, stats uniform across each 16-lane group
        float alpha[8], p0v[8], p1v[8];
#pragma unroll
        for (int r = 0; r < 8; ++r) {
            float mx = fmaxf(s0[r], s1[r]);
#pragma unroll
            for (int off = 8; off >= 1; off >>= 1) mx = fmaxf(mx, __shfl_xor(mx, off, 32));
            float mnew = fmaxf(mrun[r], mx);
            alpha[r] = __expf(mrun[r] - mnew);
            mrun[r] = mnew;
            float e0 = __expf(s0[r] - mnew), e1 = __expf(s1[r] - mnew);
            p0v[r] = e0; p1v[r] = e1;
            float rs = e0 + e1;
#pragma unroll
            for (int off = 8; off >= 1; off >>= 1) rs += __shfl_xor(rs, off, 32);
            lrun[r] = lrun[r] * alpha[r] + rs;
        }
#pragma unroll
        for (int nt = 0; nt < 8; ++nt)
#pragma unroll
            for (int r = 0; r < 8; ++r) o[nt][r] *= alpha[r];
        // write P (C-layout) into LDS already in A-frag order:
        // value at (m=r+8*hi, col) -> Pl[lane'*16+e'], lane' = ((col>>4)&1)*16+m,
        // e' = col&15; cols are mr and mr+16 (same e').
#pragma unroll
        for (int r = 0; r < 8; ++r) {
            int lp = r + (hi << 3);
            Pl[lp * 16 + mr]        = f2bf(p0v[r]);
            Pl[(lp + 16) * 16 + mr] = f2bf(p1v[r]);
        }
        __syncthreads();
        v16bf pf = *(const v16bf*)(&Pl[lane * 16]);
#pragma unroll
        for (int nt = 0; nt < 8; ++nt) {
            v16bf bv = *(const v16bf*)(Vp + (size_t)kt * 4096 + nt * 512 + lane * 16);
            o[nt] = wmma_bf(pf, bv, o[nt]);
        }
        __syncthreads();
    }
#pragma unroll
    for (int nt = 0; nt < 8; ++nt)
#pragma unroll
        for (int r = 0; r < 8; ++r) {
            int tok = qt * 16 + r + (hi << 3);
            Ob[(size_t)bn * 2048 * 512 + (size_t)tok * 512 + h * 128 + nt * 16 + mr] =
                f2bf(o[nt][r] / lrun[r]);
        }
}

// ---------------- InstanceNorm(+pos)+ReLU + im2col for conv2x2 s2 ----------------
__global__ __launch_bounds__(256) void lm3_inorm_conv(const float* __restrict__ src,
                                                      const float* __restrict__ pos,
                                                      unsigned short* __restrict__ A) {
    __shared__ float sh[256];
    int bc = blockIdx.x, bn = bc >> 7, c = bc & 127, tid = threadIdx.x;
    const float* x  = src + (size_t)bc * 4096;
    const float* pz = pos + (size_t)c * 4096;
    float s = 0.f, s2 = 0.f;
    for (int i = tid; i < 4096; i += 256) { float v = x[i] + pz[i]; s += v; s2 += v * v; }
    sh[tid] = s; __syncthreads();
    for (int st = 128; st > 0; st >>= 1) { if (tid < st) sh[tid] += sh[tid + st]; __syncthreads(); }
    float mean = sh[0] * (1.f / 4096.f); __syncthreads();
    sh[tid] = s2; __syncthreads();
    for (int st = 128; st > 0; st >>= 1) { if (tid < st) sh[tid] += sh[tid + st]; __syncthreads(); }
    float var = sh[0] * (1.f / 4096.f) - mean * mean;
    float rstd = rsqrtf(fmaxf(var, 0.f) + LM_EPS);
    for (int i = tid; i < 4096; i += 256) {
        float v = (x[i] + pz[i] - mean) * rstd;
        v = v > 0.f ? v : 0.f;
        int yy = i >> 6, xx = i & 63;
        int t = (yy >> 1) * 32 + (xx >> 1);
        int k = c * 4 + (yy & 1) * 2 + (xx & 1);
        A[((size_t)bn * 1024 + t) * 512 + k] = f2bf(v);
    }
}

// InstanceNorm+ReLU on token-major z[bn][T][128] -> bf16 A (same layout), per (bn,channel)
__global__ __launch_bounds__(256) void lm3_inorm_T(const float* __restrict__ z,
                                                   unsigned short* __restrict__ A, int T) {
    __shared__ float sh[256];
    int bc = blockIdx.x, bn = bc >> 7, d = bc & 127, tid = threadIdx.x;
    const float* x = z + (size_t)bn * T * 128 + d;
    float s = 0.f, s2 = 0.f;
    for (int t = tid; t < T; t += 256) { float v = x[(size_t)t * 128]; s += v; s2 += v * v; }
    sh[tid] = s; __syncthreads();
    for (int st = 128; st > 0; st >>= 1) { if (tid < st) sh[tid] += sh[tid + st]; __syncthreads(); }
    float mean = sh[0] / (float)T; __syncthreads();
    sh[tid] = s2; __syncthreads();
    for (int st = 128; st > 0; st >>= 1) { if (tid < st) sh[tid] += sh[tid + st]; __syncthreads(); }
    float var = sh[0] / (float)T - mean * mean;
    float rstd = rsqrtf(fmaxf(var, 0.f) + LM_EPS);
    for (int t = tid; t < T; t += 256) {
        float v = (x[(size_t)t * 128] - mean) * rstd;
        v = v > 0.f ? v : 0.f;
        A[((size_t)bn * T + t) * 128 + d] = f2bf(v);
    }
}

// ---------------- LayerNorm over 128, one wave per row ----------------
__global__ __launch_bounds__(256) void lm3_ln(const float* __restrict__ src,
                                              const float* __restrict__ g,
                                              const float* __restrict__ bb,
                                              float* __restrict__ of,
                                              unsigned short* __restrict__ ob, int rows) {
    int row = (blockIdx.x * 256 + threadIdx.x) >> 5;
    int lane = threadIdx.x & 31;
    if (row >= rows) return;
    const float* x = src + (size_t)row * 128;
    float v0 = x[lane], v1 = x[lane + 32], v2 = x[lane + 64], v3 = x[lane + 96];
    float s = v0 + v1 + v2 + v3;
#pragma unroll
    for (int off = 16; off >= 1; off >>= 1) s += __shfl_xor(s, off, 32);
    float mean = s * (1.f / 128.f);
    float d0 = v0 - mean, d1 = v1 - mean, d2 = v2 - mean, d3 = v3 - mean;
    float s2 = d0 * d0 + d1 * d1 + d2 * d2 + d3 * d3;
#pragma unroll
    for (int off = 16; off >= 1; off >>= 1) s2 += __shfl_xor(s2, off, 32);
    float rstd = rsqrtf(s2 * (1.f / 128.f) + LM_EPS);
    float y0 = d0 * rstd * g[lane]      + bb[lane];
    float y1 = d1 * rstd * g[lane + 32] + bb[lane + 32];
    float y2 = d2 * rstd * g[lane + 64] + bb[lane + 64];
    float y3 = d3 * rstd * g[lane + 96] + bb[lane + 96];
    size_t b = (size_t)row * 128;
    if (of) { of[b + lane] = y0; of[b + lane + 32] = y1; of[b + lane + 64] = y2; of[b + lane + 96] = y3; }
    if (ob) { ob[b + lane] = f2bf(y0); ob[b + lane + 32] = f2bf(y1);
              ob[b + lane + 64] = f2bf(y2); ob[b + lane + 96] = f2bf(y3); }
}

// LayerNorm of the query stream (gathers cam/lid token-major conv outputs)
__global__ __launch_bounds__(256) void lm3_lnq(const float* __restrict__ came,
                                               const float* __restrict__ lide,
                                               const float* __restrict__ g,
                                               const float* __restrict__ bb,
                                               unsigned short* __restrict__ ob) {
    int row = (blockIdx.x * 256 + threadIdx.x) >> 5;
    int lane = threadIdx.x & 31;
    if (row >= 16384) return;
    int bn = row >> 11, i = row & 2047;
    const float* x = (i < 1024) ? came + ((size_t)bn * 1024 + i) * 128
                                : lide + ((size_t)bn * 1024 + (i - 1024)) * 128;
    float v0 = x[lane], v1 = x[lane + 32], v2 = x[lane + 64], v3 = x[lane + 96];
    float s = v0 + v1 + v2 + v3;
#pragma unroll
    for (int off = 16; off >= 1; off >>= 1) s += __shfl_xor(s, off, 32);
    float mean = s * (1.f / 128.f);
    float d0 = v0 - mean, d1 = v1 - mean, d2 = v2 - mean, d3 = v3 - mean;
    float s2 = d0 * d0 + d1 * d1 + d2 * d2 + d3 * d3;
#pragma unroll
    for (int off = 16; off >= 1; off >>= 1) s2 += __shfl_xor(s2, off, 32);
    float rstd = rsqrtf(s2 * (1.f / 128.f) + LM_EPS);
    size_t b = (size_t)row * 128;
    ob[b + lane]      = f2bf(d0 * rstd * g[lane]      + bb[lane]);
    ob[b + lane + 32] = f2bf(d1 * rstd * g[lane + 32] + bb[lane + 32]);
    ob[b + lane + 64] = f2bf(d2 * rstd * g[lane + 64] + bb[lane + 64]);
    ob[b + lane + 96] = f2bf(d3 * rstd * g[lane + 96] + bb[lane + 96]);
}

// ---------------- pack / elementwise kernels ----------------
__global__ void lm3_cast(const float* __restrict__ s, unsigned short* __restrict__ d,
                         int n, float scale) {
    int i = blockIdx.x * 256 + threadIdx.x;
    if (i < n) d[i] = f2bf(s[i] * scale);
}
// f32 row-major B[k][n] (KxN) -> swizzled fragment layout
__global__ void lm3_castB(const float* __restrict__ s, unsigned short* __restrict__ d,
                          int N, int n) {
    int i = blockIdx.x * 256 + threadIdx.x;
    if (i >= n) return;
    int k = i / N, nn = i % N;
    d[bsw_idx(k, nn, N)] = f2bf(s[i]);
}
__global__ void lm3_pack_convw(const float* __restrict__ w, unsigned short* __restrict__ B) {
    int i = blockIdx.x * 256 + threadIdx.x;               // 65536: w[o][c][a][d]
    int dd = i & 1, a = (i >> 1) & 1, c = (i >> 2) & 127, o = i >> 9;
    B[bsw_idx(c * 4 + a * 2 + dd, o, 128)] = f2bf(w[i]);
}
__global__ void lm3_pack_convTw(const float* __restrict__ w, unsigned short* __restrict__ B) {
    int i = blockIdx.x * 256 + threadIdx.x;               // 65536: w[i][o][a][d]
    int dd = i & 1, a = (i >> 1) & 1, o = (i >> 2) & 127, ii = i >> 9;
    B[bsw_idx(ii, o * 4 + a * 2 + dd, 512)] = f2bf(w[i]);
}
// K raw f32 [bn*1024][512] -> Ksw B-frags of K^T per (bn,h)
__global__ void lm3_packK(const float* __restrict__ raw, unsigned short* __restrict__ Ksw) {
    size_t idx = (size_t)blockIdx.x * 256 + threadIdx.x;  // 8*1024*512
    int col = (int)(idx & 511);
    int row = (int)(idx >> 9);
    int bn = row >> 10, tok = row & 1023;
    int h = col >> 7, kd = col & 127;
    int lane = (((kd >> 4) & 1) << 4) + (tok & 15);
    size_t dst = ((((size_t)(bn * 4 + h) * 4 + (kd >> 5)) * 64 + (tok >> 4)) * 32 + lane) * 16
                 + (kd & 15);
    Ksw[dst] = f2bf(raw[idx]);
}
// V raw f32 [bn*1024][512] -> Vsw B-frags of V per (bn,h)
__global__ void lm3_packV(const float* __restrict__ raw, unsigned short* __restrict__ Vsw) {
    size_t idx = (size_t)blockIdx.x * 256 + threadIdx.x;  // 8*1024*512
    int col = (int)(idx & 511);
    int row = (int)(idx >> 9);
    int bn = row >> 10, tok = row & 1023;
    int h = col >> 7, vd = col & 127;
    int lane = (((tok >> 4) & 1) << 4) + (vd & 15);
    size_t dst = ((((size_t)(bn * 4 + h) * 32 + (tok >> 5)) * 8 + (vd >> 4)) * 32 + lane) * 16
                 + (tok & 15);
    Vsw[dst] = f2bf(raw[idx]);
}
// z0 = raw + proj_b + skip   (skip: cat along width of cam_e/lid_e)
__global__ void lm3_bias_skip(const float* __restrict__ raw, const float* __restrict__ pb,
                              const float* __restrict__ came, const float* __restrict__ lide,
                              float* __restrict__ z0) {
    size_t idx = (size_t)blockIdx.x * 256 + threadIdx.x;  // 16384*128
    int d = idx & 127;
    size_t row = idx >> 7;
    int bn = (int)(row >> 11), i = (int)(row & 2047);
    int hh = i >> 6, ww = i & 63;
    float sk = (ww < 32) ? came[((size_t)bn * 1024 + hh * 32 + ww) * 128 + d]
                         : lide[((size_t)bn * 1024 + hh * 32 + (ww - 32)) * 128 + d];
    z0[idx] = raw[idx] + pb[d] + sk;
}
__global__ void lm3_gelu(const float* __restrict__ raw, const float* __restrict__ b1,
                         unsigned short* __restrict__ hb) {
    size_t idx = (size_t)blockIdx.x * 256 + threadIdx.x;  // 16384*256
    float x = raw[idx] + b1[idx & 255];
    hb[idx] = f2bf(0.5f * x * (1.f + erff(x * 0.70710678118654752f)));
}
__global__ void lm3_bias_res(const float* __restrict__ raw, const float* __restrict__ b2,
                             const float* __restrict__ z1, float* __restrict__ z2) {
    size_t idx = (size_t)blockIdx.x * 256 + threadIdx.x;  // 16384*128
    z2[idx] = z1[idx] + raw[idx] + b2[idx & 127];
}
// fused2[bn][hh*32+ww][d] = (zc+zl)[y=hh] + (zc+zl)[y=hh+32]
__global__ void lm3_combine(const float* __restrict__ zc, const float* __restrict__ zl,
                            float* __restrict__ f2) {
    size_t idx = (size_t)blockIdx.x * 256 + threadIdx.x;  // 8*1024*128
    int d = idx & 127;
    int t = (int)((idx >> 7) & 1023);
    int bn = (int)(idx >> 17);
    int hh = t >> 5, ww = t & 31;
    size_t base = (size_t)bn * 2048 * 128;
    size_t i0 = base + (size_t)(hh * 32 + ww) * 128 + d;
    size_t i1 = base + (size_t)((hh + 32) * 32 + ww) * 128 + d;
    f2[idx] = zc[i0] + zl[i0] + zc[i1] + zl[i1];
}
// convT scatter: raw[token][o*4+a*2+d] -> out[bn][o][2h+a][2w+d]
__global__ void lm3_scat(const float* __restrict__ raw, float* __restrict__ out,
                         int T, int Hin, int Win) {
    size_t idx = (size_t)blockIdx.x * 256 + threadIdx.x;  // 8*T*512 exact
    int token = (int)(idx >> 9), q = (int)(idx & 511);
    int o = q >> 2, a = (q >> 1) & 1, dd = q & 1;
    int bn = token / T, p = token % T;
    int hh = p / Win, ww = p % Win;
    int Hout = Hin * 2, Wout = Win * 2;
    out[((size_t)bn * 128 + o) * Hout * Wout + (size_t)(2 * hh + a) * Wout + (2 * ww + dd)] = raw[idx];
}

// ---------------- host orchestration ----------------
static void launch_gemm(const unsigned short* A, const unsigned short* Bsw, float* C,
                        int M, int N, int K, hipStream_t s) {
    dim3 g(M >> 7, N >> 6);
    lm3_gemm<<<g, 256, 0, s>>>(A, Bsw, C, M, N, K);
}

extern "C" void kernel_launch(void* const* d_in, const int* in_sizes, int n_in,
                              void* d_out, int out_size, void* d_ws, size_t ws_size,
                              hipStream_t stream) {
    (void)in_sizes; (void)n_in; (void)out_size; (void)ws_size;
    const float* cam_feat = (const float*)d_in[0];
    const float* lid_feat = (const float*)d_in[1];
    const float* pos      = (const float*)d_in[2];
    const float* conv_w   = (const float*)d_in[3];
    const float* convT_w  = (const float*)d_in[4];
    const float* lnq_g = (const float*)d_in[5],  *lnq_b = (const float*)d_in[6];
    const float* wq    = (const float*)d_in[7];
    const float* lnk_g = (const float*)d_in[8],  *lnk_b = (const float*)d_in[9];
    const float* wk    = (const float*)d_in[10];
    const float* lnv_g = (const float*)d_in[11], *lnv_b = (const float*)d_in[12];
    const float* wv    = (const float*)d_in[13];
    const float* proj_w = (const float*)d_in[14], *proj_b = (const float*)d_in[15];
    const float* pre_g  = (const float*)d_in[16], *pre_b  = (const float*)d_in[17];
    const float* m1w = (const float*)d_in[18], *m1bv = (const float*)d_in[19];
    const float* m2w = (const float*)d_in[20], *m2bv = (const float*)d_in[21];
    const float* post_g = (const float*)d_in[22], *post_b = (const float*)d_in[23];
    float* out = (float*)d_out;
    char* ws = (char*)d_ws;
    const size_t MB = 1ull << 20;

    unsigned short* Acam  = (unsigned short*)(ws + 0 * MB);      // 8 MB
    unsigned short* Alid  = (unsigned short*)(ws + 8 * MB);      // 8 MB
    unsigned short* Bconv = (unsigned short*)(ws + 16 * MB);
    unsigned short* BT    = (unsigned short*)(ws + 16 * MB + 256 * 1024);
    unsigned short* wqb   = (unsigned short*)(ws + 16 * MB + 512 * 1024);
    unsigned short* wkb   = (unsigned short*)(ws + 16 * MB + 768 * 1024);
    unsigned short* wvb   = (unsigned short*)(ws + 17 * MB);
    unsigned short* pwb   = (unsigned short*)(ws + 17 * MB + 256 * 1024);
    unsigned short* m1b   = (unsigned short*)(ws + 17 * MB + 512 * 1024);
    unsigned short* m2b   = (unsigned short*)(ws + 17 * MB + 768 * 1024);
    float*          cam_e = (float*)(ws + 18 * MB);              // 4 MB
    float*          lid_e = (float*)(ws + 22 * MB);              // 4 MB
    unsigned short* Aq    = (unsigned short*)(ws + 26 * MB);     // 4 MB
    unsigned short* Ak    = (unsigned short*)(ws + 30 * MB);     // 2 MB
    unsigned short* Av    = (unsigned short*)(ws + 32 * MB);     // 2 MB
    unsigned short* Qbf   = (unsigned short*)(ws + 34 * MB);     // 16 MB
    unsigned short* Ksw   = (unsigned short*)(ws + 50 * MB);     // 8 MB
    unsigned short* Vsw   = (unsigned short*)(ws + 58 * MB);     // 8 MB
    unsigned short* Aatt  = (unsigned short*)(ws + 66 * MB);     // 16 MB
    float*          raw   = (float*)(ws + 82 * MB);              // 32 MB
    float*          z0    = (float*)(ws + 114 * MB);             // 8 MB
    float*          z1f   = (float*)(ws + 122 * MB);             // 8 MB
    unsigned short* z1b   = (unsigned short*)(ws + 130 * MB);    // 4 MB
    unsigned short* Hbf   = (unsigned short*)(ws + 134 * MB);    // 8 MB
    float*          z2    = (float*)(ws + 142 * MB);             // 8 MB
    float*          zcam  = (float*)(ws + 150 * MB);             // 8 MB
    float*          zlid  = (float*)(ws + 158 * MB);             // 8 MB
    float*          fus2  = (float*)(ws + 166 * MB);             // 4 MB
    unsigned short* AT    = (unsigned short*)(ws + 170 * MB);    // 4 MB

    // weight packing into swizzled fragment layout
    lm3_pack_convw <<<256, 256, 0, stream>>>(conv_w,  Bconv);
    lm3_pack_convTw<<<256, 256, 0, stream>>>(convT_w, BT);
    lm3_castB<<<256, 256, 0, stream>>>(wq,     wqb, 512, 65536);
    lm3_castB<<<256, 256, 0, stream>>>(wk,     wkb, 512, 65536);
    lm3_castB<<<256, 256, 0, stream>>>(wv,     wvb, 512, 65536);
    lm3_castB<<<256, 256, 0, stream>>>(proj_w, pwb, 128, 65536);
    lm3_castB<<<128, 256, 0, stream>>>(m1w,    m1b, 256, 32768);
    lm3_castB<<<128, 256, 0, stream>>>(m2w,    m2b, 128, 32768);

    // stage 1: +pos, inorm, relu, conv2x2s2 as GEMM
    lm3_inorm_conv<<<1024, 256, 0, stream>>>(cam_feat, pos, Acam);
    lm3_inorm_conv<<<1024, 256, 0, stream>>>(lid_feat, pos, Alid);
    launch_gemm(Acam, Bconv, cam_e, 8192, 128, 512, stream);
    launch_gemm(Alid, Bconv, lid_e, 8192, 128, 512, stream);

    // queries (shared by both passes); DH^-0.5 folded into Q
    lm3_lnq<<<2048, 256, 0, stream>>>(cam_e, lid_e, lnq_g, lnq_b, Aq);
    launch_gemm(Aq, wqb, raw, 16384, 512, 128, stream);
    lm3_cast<<<32768, 256, 0, stream>>>(raw, Qbf, 16384 * 512, 0.08838834764831845f);

    for (int pass = 0; pass < 2; ++pass) {
        const float* kv = pass ? lid_e : cam_e;
        lm3_ln<<<1024, 256, 0, stream>>>(kv, lnk_g, lnk_b, nullptr, Ak, 8192);
        launch_gemm(Ak, wkb, raw, 8192, 512, 128, stream);
        lm3_packK<<<16384, 256, 0, stream>>>(raw, Ksw);
        lm3_ln<<<1024, 256, 0, stream>>>(kv, lnv_g, lnv_b, nullptr, Av, 8192);
        launch_gemm(Av, wvb, raw, 8192, 512, 128, stream);
        lm3_packV<<<16384, 256, 0, stream>>>(raw, Vsw);

        lm3_attn<<<dim3(128, 4, 8), 32, 0, stream>>>(Qbf, Ksw, Vsw, Aatt);

        launch_gemm(Aatt, pwb, raw, 16384, 128, 512, stream);
        lm3_bias_skip<<<8192, 256, 0, stream>>>(raw, proj_b, cam_e, lid_e, z0);
        lm3_ln<<<2048, 256, 0, stream>>>(z0, pre_g, pre_b, z1f, z1b, 16384);
        launch_gemm(z1b, m1b, raw, 16384, 256, 128, stream);
        lm3_gelu<<<16384, 256, 0, stream>>>(raw, m1bv, Hbf);
        launch_gemm(Hbf, m2b, raw, 16384, 128, 256, stream);
        lm3_bias_res<<<8192, 256, 0, stream>>>(raw, m2bv, z1f, z2);
        lm3_ln<<<2048, 256, 0, stream>>>(z2, post_g, post_b, pass ? zlid : zcam, nullptr, 16384);
    }

    // fused = (cam+lid) folded over the (2, Y/2) split
    lm3_combine<<<4096, 256, 0, stream>>>(zcam, zlid, fus2);

    // convT branches: inorm+relu, GEMM vs packed convT weights, scatter to output
    lm3_inorm_T<<<1024, 256, 0, stream>>>(fus2, AT, 1024);
    launch_gemm(AT, BT, raw, 8192, 512, 128, stream);
    lm3_scat<<<16384, 256, 0, stream>>>(raw, out, 1024, 32, 32);

    lm3_inorm_T<<<1024, 256, 0, stream>>>(zcam, AT, 2048);
    launch_gemm(AT, BT, raw, 16384, 512, 128, stream);
    lm3_scat<<<32768, 256, 0, stream>>>(raw, out + 4194304, 2048, 64, 32);

    lm3_inorm_T<<<1024, 256, 0, stream>>>(zlid, AT, 2048);
    launch_gemm(AT, BT, raw, 16384, 512, 128, stream);
    lm3_scat<<<32768, 256, 0, stream>>>(raw, out + 12582912, 2048, 64, 32);
}
